// RNN_2250562863460
// MI455X (gfx1250) — matrix operand order repaired and checked by reading, not compiled, not run on hardware
//
#include <hip/hip_runtime.h>
#include <hip/hip_bf16.h>

#define T_STEPS 28
#define IN_DIM  28
#define HID     64
#define OUT_DIM 10

typedef __attribute__((ext_vector_type(16))) _Float16 v16h;
typedef __attribute__((ext_vector_type(8)))  _Float16 h8;
typedef __attribute__((ext_vector_type(8)))  float    v8f;

__device__ __forceinline__ v16h ld_frag(const _Float16* p0, const _Float16* p1) {
    // Build a 16xf16 WMMA operand fragment from two 16-byte LDS runs.
    h8 lo = *(const h8*)p0;
    h8 hi = *(const h8*)p1;
    v16h r;
#pragma unroll
    for (int i = 0; i < 8; ++i) { r[i] = lo[i]; r[8 + i] = hi[i]; }
    return r;
}

// gfx1250 hardware v_tanh_f32 (single TRANS op per activation).
#if __has_builtin(__builtin_amdgcn_tanhf)
__device__ __forceinline__ float tanh_f(float x) { return __builtin_amdgcn_tanhf(x); }
#else
__device__ __forceinline__ float tanh_f(float x) {
    return 1.0f - 2.0f * __builtin_amdgcn_rcpf(1.0f + __expf(2.0f * x));
}
#endif
// Pre-scaled sigmoid: weights/bias for i,f,o gates carry the 0.5 factor, so
// sigmoid(orig) = 0.5*tanh(acc) + 0.5 with acc = 0.5*orig already.
__device__ __forceinline__ float sigmoid_ps(float acc) {
    return __builtin_fmaf(0.5f, tanh_f(acc), 0.5f);
}

__device__ __forceinline__ v8f wmma_f16(v16h a, v16h b, v8f c) {
    // (neg_a, A, neg_b, B, c_mod, C, reuse_a, reuse_b)
    return __builtin_amdgcn_wmma_f32_16x16x32_f16(false, a, false, b, (short)0, c, false, false);
}

// Gate rows: i:[0,64) f:[64,128) g:[128,192) o:[192,256). Sigmoid gates scaled by 0.5.
__device__ __forceinline__ float gate_scale(int r) {
    return (r >= 128 && r < 192) ? 1.0f : 0.5f;
}

__global__ __launch_bounds__(64) void lstm_fused_kernel(
    const float* __restrict__ x,      // [B, T, I]
    const float* __restrict__ W_ih,   // [4H, I]
    const float* __restrict__ W_hh,   // [4H, H]
    const float* __restrict__ b_ih,   // [4H]
    const float* __restrict__ b_hh,   // [4H]
    const float* __restrict__ W_out,  // [O, H]
    const float* __restrict__ b_out,  // [O]
    float* __restrict__ out)          // [B, O]
{
    // Weights in f16, row-major [N][K] so B-operand fragments are contiguous 16B LDS reads.
    __shared__ _Float16 s_wih[256 * 32];   // K padded 28 -> 32 (i/f/o rows pre-scaled 0.5)
    __shared__ _Float16 s_whh[256 * 64];   // (i/f/o rows pre-scaled 0.5)
    __shared__ float    s_bias[256];       // (i/f/o pre-scaled 0.5)
    __shared__ _Float16 s_wout[16 * 64];   // O padded 10 -> 16
    __shared__ float    s_bout[16];
    __shared__ _Float16 s_h[2][32 * 64];   // per-wave h tile, row-major [M=32][K=64]

    const int tid  = threadIdx.x;
    const int wave = tid >> 5;
    const int lane = tid & 31;
    const int col  = lane & 15;                 // N (and M for A-fragment rows)
    const int hsel = (lane & 16) ? 1 : 0;       // which half of the lane group

    // ---- cooperative LDS init (64 threads) ----
    for (int i = tid; i < 256 * 32; i += 64) {
        int r = i >> 5, c = i & 31;
        s_wih[i] = (c < IN_DIM) ? (_Float16)(gate_scale(r) * W_ih[r * IN_DIM + c])
                                : (_Float16)0.0f;
    }
    for (int i = tid; i < 256 * 64; i += 64) {
        int r = i >> 6;
        s_whh[i] = (_Float16)(gate_scale(r) * W_hh[i]);
    }
    for (int i = tid; i < 256; i += 64)
        s_bias[i] = gate_scale(i) * (b_ih[i] + b_hh[i]);
    for (int i = tid; i < 16 * 64; i += 64) {
        int r = i >> 6, c = i & 63;
        s_wout[i] = (r < OUT_DIM) ? (_Float16)W_out[r * HID + c] : (_Float16)0.0f;
    }
    for (int i = tid; i < 16; i += 64)        s_bout[i] = (i < OUT_DIM) ? b_out[i] : 0.0f;
    for (int i = tid; i < 2 * 32 * 64; i += 64) (&s_h[0][0])[i] = (_Float16)0.0f;
    __syncthreads();

    const int rowbase = blockIdx.x * 64 + wave * 32;   // this wave's 32 batch rows
    _Float16* hbuf = s_h[wave];

    // Per-lane bias (one value per N-tile): loop-invariant, keep in registers.
    float bias_r[16];
#pragma unroll
    for (int n = 0; n < 16; ++n) bias_r[n] = s_bias[n * 16 + col];

    float c_state[2][4][8];
#pragma unroll
    for (int mt = 0; mt < 2; ++mt)
#pragma unroll
        for (int m = 0; m < 4; ++m)
#pragma unroll
            for (int v = 0; v < 8; ++v) c_state[mt][m][v] = 0.0f;

    const int k0 = hsel ? 16 : 0;  // A-fragment K offset (contiguous order k = 16h + s)
    const float* xbase[2];
#pragma unroll
    for (int mt = 0; mt < 2; ++mt)
        xbase[mt] = x + (size_t)(rowbase + mt * 16 + col) * (T_STEPS * IN_DIM);

    for (int t = 0; t < T_STEPS; ++t) {
        // A fragments of x_t for both 16-row tiles (16x32, K padded 28->32)
        v16h a_x[2];
        v16h a_hlo[2], a_hhi[2];
#pragma unroll
        for (int mt = 0; mt < 2; ++mt) {
            const float* xp = xbase[mt] + t * IN_DIM;
            if (t + 1 < T_STEPS) __builtin_prefetch(xp + IN_DIM, 0, 0);
#pragma unroll
            for (int i = 0; i < 8; ++i) a_x[mt][i] = (_Float16)xp[k0 + i];
#pragma unroll
            for (int i = 0; i < 8; ++i) {
                int k = k0 + 8 + i;
                a_x[mt][8 + i] = (k < IN_DIM) ? (_Float16)xp[k] : (_Float16)0.0f;
            }
            // A fragments of h (16x64 = two 16x32 halves), from row-major LDS
            const _Float16* hp = hbuf + (mt * 16 + col) * 64;
            a_hlo[mt] = ld_frag(hp + k0,      hp + 8 + k0);
            a_hhi[mt] = ld_frag(hp + 32 + k0, hp + 40 + k0);
        }

#pragma unroll
        for (int m = 0; m < 4; ++m) {            // hidden-unit chunk (16 wide)
            v8f acc[4][2];
#pragma unroll
            for (int g = 0; g < 4; ++g) {        // gate: i, f, g, o
                const int n = g * 4 + m;         // N-tile index in [0,16)
                // Load each B fragment ONCE, feed both M-tiles (2x WMMA reuse).
                const _Float16* bp_ih = s_wih + (n * 16 + col) * 32 + (hsel ? 16 : 0);
                v16h b_x = ld_frag(bp_ih, bp_ih + 8);
                const _Float16* bp0 = s_whh + (n * 16 + col) * 64 + (hsel ? 16 : 0);
                v16h b_h0 = ld_frag(bp0, bp0 + 8);
                const _Float16* bp1 = bp0 + 32;
                v16h b_h1 = ld_frag(bp1, bp1 + 8);

                const float bb = bias_r[n];
#pragma unroll
                for (int mt = 0; mt < 2; ++mt) {
                    v8f a = { bb, bb, bb, bb, bb, bb, bb, bb };
                    a = wmma_f16(a_x[mt],   b_x,  a);
                    a = wmma_f16(a_hlo[mt], b_h0, a);
                    a = wmma_f16(a_hhi[mt], b_h1, a);
                    acc[g][mt] = a;
                }
            }
#pragma unroll
            for (int mt = 0; mt < 2; ++mt) {
#pragma unroll
                for (int v = 0; v < 8; ++v) {
                    float ig = sigmoid_ps(acc[0][mt][v]);
                    float fg = sigmoid_ps(acc[1][mt][v]);
                    float gg = tanh_f    (acc[2][mt][v]);
                    float og = sigmoid_ps(acc[3][mt][v]);
                    float cn = fg * c_state[mt][m][v] + ig * gg;
                    c_state[mt][m][v] = cn;
                    float hn = og * tanh_f(cn);
                    int M = mt * 16 + v + hsel * 8;  // C-layout row for this lane/VGPR
                    hbuf[M * 64 + m * 16 + col] = (_Float16)hn;
                }
            }
        }
    }

    // ---- output projection: h_T [32x64] @ W_out^T [64x16(pad)] + b_out ----
    {
        const _Float16* bp0 = s_wout + col * 64 + (hsel ? 16 : 0);
        const _Float16* bp1 = bp0 + 32;
        v16h b0 = ld_frag(bp0, bp0 + 8);
        v16h b1 = ld_frag(bp1, bp1 + 8);
        const float bb = s_bout[col];
#pragma unroll
        for (int mt = 0; mt < 2; ++mt) {
            const _Float16* hp = hbuf + (mt * 16 + col) * 64;
            v16h a_hlo = ld_frag(hp + k0,      hp + 8 + k0);
            v16h a_hhi = ld_frag(hp + 32 + k0, hp + 40 + k0);
            v8f acc = { bb, bb, bb, bb, bb, bb, bb, bb };
            acc = wmma_f16(a_hlo, b0, acc);
            acc = wmma_f16(a_hhi, b1, acc);
            if (col < OUT_DIM) {
#pragma unroll
                for (int v = 0; v < 8; ++v) {
                    int M = mt * 16 + v + hsel * 8;
                    out[(size_t)(rowbase + M) * OUT_DIM + col] = acc[v];
                }
            }
        }
    }
}

extern "C" void kernel_launch(void* const* d_in, const int* in_sizes, int n_in,
                              void* d_out, int out_size, void* d_ws, size_t ws_size,
                              hipStream_t stream) {
    (void)n_in; (void)out_size; (void)d_ws; (void)ws_size;
    const float* x     = (const float*)d_in[0];
    const float* W_ih  = (const float*)d_in[1];
    const float* W_hh  = (const float*)d_in[2];
    const float* b_ih  = (const float*)d_in[3];
    const float* b_hh  = (const float*)d_in[4];
    const float* W_out = (const float*)d_in[5];
    const float* b_out = (const float*)d_in[6];
    float* out = (float*)d_out;

    const int B = in_sizes[0] / (T_STEPS * IN_DIM);   // 65536
    const int blocks = (B + 63) / 64;                 // 64 batch rows per block (2 waves x 32)
    lstm_fused_kernel<<<blocks, 64, 0, stream>>>(x, W_ih, W_hh, b_ih, b_hh, W_out, b_out, out);
}
